// PyTorchConv1d_2396591751363
// MI455X (gfx1250) — matrix operand order repaired and checked by reading, not compiled, not run on hardware
//
#include <hip/hip_runtime.h>

typedef __attribute__((ext_vector_type(16))) __bf16 v16bf;
typedef __attribute__((ext_vector_type(8)))  float  v8f;
typedef __attribute__((ext_vector_type(4)))  unsigned int u32x4;

namespace {

constexpr int N_IMG  = 32;
constexpr int C_IN   = 64;
constexpr int W_IN   = 4096;
constexpr int F_OUT  = 128;
constexpr int KW     = 64;
constexpr int OUT_W  = W_IN - KW + 1;   // 4033
constexpr int TILE_T = 256;             // output positions per block
constexpr int KSTEP  = 32;              // K per WMMA (bf16 16x16x32)
constexpr int NSUB   = TILE_T / 16;     // 16 N-subtiles per wave
constexpr int XWIN   = 296;             // staged x window (halves), 592B stride
constexpr int AROW   = 40;              // padded LDS row stride for w slice (halves)
constexpr int NKS    = (C_IN * KW) / KSTEP;  // 128 K-steps

union FragAB {
  v16bf v;
  u32x4 q[2];
};

__device__ __forceinline__ unsigned short f2bf(float f) {
  unsigned int u = __builtin_bit_cast(unsigned int, f);
  u += 0x7fffu + ((u >> 16) & 1u);   // round-to-nearest-even
  return (unsigned short)(u >> 16);
}

} // namespace

__global__ __launch_bounds__(256) void conv1d_wmma_bf16_pipe(
    const float* __restrict__ x, const float* __restrict__ w,
    const float* __restrict__ bias, float* __restrict__ out) {
  // Double-buffered LDS stages (software pipeline, one barrier per K-step).
  __shared__ alignas(16) unsigned short As[2][F_OUT * AROW];  // bf16 w slice
  __shared__ alignas(16) unsigned short Xc[2][8 * XWIN];      // 8 shifted x copies

  const int tid   = threadIdx.x;
  const int lane  = tid & 31;
  const int wave  = tid >> 5;
  const int r     = lane & 15;
  const int hi    = lane >> 4;       // 0 | 1 (lane half)
  const int kbA   = hi * 16;         // A-frag K base (contiguous K per lane half, matches B)
  const int kbB   = hi * 16;         // B-frag K base (B rows striped across lanes)
  const int fbase = wave * 16;       // 16-filter strip per wave

  const int t0 = blockIdx.x * TILE_T;
  const int n  = blockIdx.y;

  const int fW   = tid >> 1;         // filter row this thread stages
  const int part = tid & 1;          // which 16-half chunk of the 32-wide slice

  const float* xn   = x + (size_t)n * C_IN * W_IN;
  const float* wrow = w + (size_t)fW * C_IN * KW + part * 16;

  v8f acc[NSUB] = {};

  // Register staging for the next K-step (global loads overlap compute).
  float4 wst[4];
  float  xst0, xst1;

  auto load_stage = [&](int ks) {
    const int c  = ks >> 1;
    const int k0 = (ks & 1) * KSTEP;
    const float4* wv = (const float4*)(wrow + c * KW + k0);
    wst[0] = wv[0]; wst[1] = wv[1]; wst[2] = wv[2]; wst[3] = wv[3];
    const float* xp = xn + (size_t)c * W_IN + t0 + k0;
    {
      const int g = t0 + k0 + tid;
      xst0 = (g < W_IN) ? xp[tid] : 0.0f;
    }
    {
      const int i = tid + 256;
      const int g = t0 + k0 + i;
      xst1 = (i < XWIN && g < W_IN) ? xp[i] : 0.0f;
    }
  };

  auto store_stage = [&](int buf) {
    // w slice -> As[buf]
    union { unsigned short h[16]; u32x4 q[2]; } pk;
    #pragma unroll
    for (int q4 = 0; q4 < 4; ++q4) {
      pk.h[q4 * 4 + 0] = f2bf(wst[q4].x);
      pk.h[q4 * 4 + 1] = f2bf(wst[q4].y);
      pk.h[q4 * 4 + 2] = f2bf(wst[q4].z);
      pk.h[q4 * 4 + 3] = f2bf(wst[q4].w);
    }
    u32x4* dst = (u32x4*)&As[buf][fW * AROW + part * 16];
    dst[0] = pk.q[0];
    dst[1] = pk.q[1];
    // x window -> 8 shift-by-j copies: Xc[buf][j][i] = bf16(xwin[i + j])
    const unsigned short hv0 = f2bf(xst0);
    #pragma unroll
    for (int j = 0; j < 8; ++j) {
      const int idx = tid - j;
      if (idx >= 0) Xc[buf][j * XWIN + idx] = hv0;
    }
    if (tid + 256 < XWIN) {
      const unsigned short hv1 = f2bf(xst1);
      #pragma unroll
      for (int j = 0; j < 8; ++j) {
        Xc[buf][j * XWIN + (tid + 256 - j)] = hv1;
      }
    }
  };

  // ---- prologue: stage 0 into buffer 0
  load_stage(0);
  store_stage(0);
  __syncthreads();

  for (int ks = 0; ks < NKS; ++ks) {
    const int  cur  = ks & 1;
    const bool more = (ks + 1) < NKS;

    // Issue next stage's global loads now; they complete during the WMMAs.
    if (more) load_stage(ks + 1);

    // ---- compute from buffer `cur`
    FragAB a;
    {
      const unsigned short* arow = &As[cur][(fbase + r) * AROW];
      a.q[0] = *(const u32x4*)&arow[kbA];        // K = kb .. kb+7
      a.q[1] = *(const u32x4*)&arow[kbA + 8];    // K = kb+8 .. kb+15
    }
    const unsigned short* xrow = &Xc[cur][(lane & 7) * XWIN + kbB + (r & 8)];

    FragAB bc, bn;
    bc.q[0] = ((const u32x4*)xrow)[0];
    bc.q[1] = ((const u32x4*)xrow)[1];
    #pragma unroll
    for (int s = 0; s < NSUB; ++s) {
      if (s + 1 < NSUB) {  // prefetch next B frag while WMMA s executes
        const u32x4* bp = (const u32x4*)&xrow[(s + 1) * 16];
        bn.q[0] = bp[0];
        bn.q[1] = bp[1];
      }
      acc[s] = __builtin_amdgcn_wmma_f32_16x16x32_bf16(
          /*neg_a=*/false, a.v, /*neg_b=*/false, bc.v,
          /*c_mod=*/(short)0, acc[s], /*reuse_a=*/false, /*reuse_b=*/false);
      bc = bn;
    }

    // Convert + store next stage into the other buffer, then single barrier.
    if (more) store_stage(cur ^ 1);
    __syncthreads();
  }

  // ---- epilogue: D 16x16 layout: lanes 0-15 -> M=v, lanes 16-31 -> M=8+v; col=r
  float bv[8];
  #pragma unroll
  for (int v = 0; v < 8; ++v) bv[v] = bias[fbase + hi * 8 + v];

  #pragma unroll
  for (int s = 0; s < NSUB; ++s) {
    const int t = t0 + s * 16 + r;
    if (t < OUT_W) {
      #pragma unroll
      for (int v = 0; v < 8; ++v) {
        const int f = fbase + hi * 8 + v;
        out[(size_t)(n * F_OUT + f) * OUT_W + t] = acc[s][v] + bv[v];
      }
    }
  }
}

extern "C" void kernel_launch(void* const* d_in, const int* in_sizes, int n_in,
                              void* d_out, int out_size, void* d_ws, size_t ws_size,
                              hipStream_t stream) {
  (void)in_sizes; (void)n_in; (void)d_ws; (void)ws_size; (void)out_size;
  const float* x = (const float*)d_in[0];   // [32, 64, 4096] f32
  const float* w = (const float*)d_in[1];   // [128, 64, 64]  f32
  const float* b = (const float*)d_in[2];   // [128]          f32
  float* out     = (float*)d_out;           // [32, 128, 4033] f32

  dim3 grid((OUT_W + TILE_T - 1) / TILE_T, N_IMG);  // 16 x 32 blocks
  conv1d_wmma_bf16_pipe<<<grid, 256, 0, stream>>>(x, w, b, out);
}
